// DQN_GNN_53910429499419
// MI455X (gfx1250) — matrix operand order repaired and checked by reading, not compiled, not run on hardware
//
#include <hip/hip_runtime.h>
#include <hip/hip_bf16.h>
#include <math.h>

// ---------------- problem constants ----------------
#define D      352
#define NN     50000
#define NE     400000
#define INDIM  2000
#define FFD    2048
#define KT     11      // D / 32  (k-tiles per row)
#define OT     22      // D / 16  (output col tiles)
#define MTILES 3125    // NN / 16 (exact)
#define GWAVES 8       // waves per GEMM block

typedef __attribute__((ext_vector_type(16))) __bf16 v16bf;
typedef __attribute__((ext_vector_type(8)))  float  v8f;

// ---------------- d_in index map (pytree: dict keys sorted) ----------------
// 0 state, 1 scores, 2 edge_index, 3 items_ready_to_cache, then params:
// attn1{in{b,w},out{b,w}} 4..7 ; attn2 8..11
// convs[l]{b,ln.b,ln.g,w} -> 12+4l .. 15+4l  (l=0..5)
// enc[l]{attn.in.b, attn.in.w, attn.out.b, attn.out.w, ff1.b, ff1.w,
//        ff2.b, ff2.w, ln1.b, ln1.g, ln2.b, ln2.g} -> 36+12l ..
// mlp1.b 72, mlp1.w 73, mlp2.b 74, mlp2.w 75, proj.b 76, proj.w 77
#define IDX_A1      4
#define IDX_A2      8
#define IDX_CONV(l) (12 + 4*(l))
#define IDX_ENC(l)  (36 + 12*(l))
#define IDX_MLP1B   72
#define IDX_MLP1W   73
#define IDX_MLP2B   74
#define IDX_MLP2W   75
#define IDX_PROJB   76
#define IDX_PROJW   77

// ---------------- front-end (seq len == 1 -> MHA == V path) ----------------
struct EncP { const float *inb,*inw,*outb,*outw,*f1b,*f1w,*f2b,*f2w,*l1b,*l1g,*l2b,*l2g; };
struct FrontP {
  const float* state;
  const float *projb,*projw;
  const float *a1_inb,*a1_inw,*a1_outb,*a1_outw;
  const float *a2_inb,*a2_inw,*a2_outb,*a2_outw;
  EncP  enc[3];
  float *h, *attn1v, *scal;
};

__device__ __forceinline__ void gemv_sh(const float* __restrict__ W, const float* __restrict__ b,
                                        const float* xs, float* os, int outd, int ind,
                                        int tid, int nt) {
  for (int o = tid; o < outd; o += nt) {
    float acc = b[o];
    const float* wr = W + (size_t)o * ind;
    for (int i = 0; i < ind; ++i) acc = fmaf(xs[i], wr[i], acc);
    os[o] = acc;
  }
}

__global__ __launch_bounds__(512) void frontend_kernel(FrontP P) {
  __shared__ float s_state[INDIM];
  __shared__ float s_t[D], s_v[D], s_o[D], s_a1[D];
  __shared__ float s_ff[FFD];
  __shared__ float red[2];
  const int tid = threadIdx.x, nt = blockDim.x;

  for (int i = tid; i < INDIM; i += nt) s_state[i] = P.state[i];
  __syncthreads();
  gemv_sh(P.projw, P.projb, s_state, s_t, D, INDIM, tid, nt);      // x = state @ projW.T + b
  __syncthreads();

  // attn1 = (x @ Wv.T + bv) @ Wout.T + bout
  gemv_sh(P.a1_inw + (size_t)2*D*D, P.a1_inb + 2*D, s_t, s_v, D, D, tid, nt);
  __syncthreads();
  gemv_sh(P.a1_outw, P.a1_outb, s_v, s_o, D, D, tid, nt);
  __syncthreads();
  for (int i = tid; i < D; i += nt) { s_a1[i] = s_o[i]; s_t[i] = s_o[i]; P.attn1v[i] = s_o[i]; }
  __syncthreads();

  for (int l = 0; l < 3; ++l) {
    const EncP e = P.enc[l];
    gemv_sh(e.inw + (size_t)2*D*D, e.inb + 2*D, s_t, s_v, D, D, tid, nt);
    __syncthreads();
    gemv_sh(e.outw, e.outb, s_v, s_o, D, D, tid, nt);
    __syncthreads();
    for (int i = tid; i < D; i += nt) s_o[i] += s_t[i];             // residual
    __syncthreads();
    if (tid == 0) { float s=0.f,s2=0.f; for (int i=0;i<D;++i){float x=s_o[i]; s+=x; s2+=x*x;}
      float m=s/D; red[0]=m; red[1]=rsqrtf(fmaxf(s2/D-m*m,0.f)+1e-5f); }
    __syncthreads();
    for (int i = tid; i < D; i += nt) s_t[i] = (s_o[i]-red[0])*red[1]*e.l1g[i] + e.l1b[i];
    __syncthreads();
    gemv_sh(e.f1w, e.f1b, s_t, s_ff, FFD, D, tid, nt);
    __syncthreads();
    for (int i = tid; i < FFD; i += nt) s_ff[i] = fmaxf(s_ff[i], 0.f);
    __syncthreads();
    gemv_sh(e.f2w, e.f2b, s_ff, s_o, D, FFD, tid, nt);
    __syncthreads();
    for (int i = tid; i < D; i += nt) s_o[i] += s_t[i];             // residual
    __syncthreads();
    if (tid == 0) { float s=0.f,s2=0.f; for (int i=0;i<D;++i){float x=s_o[i]; s+=x; s2+=x*x;}
      float m=s/D; red[0]=m; red[1]=rsqrtf(fmaxf(s2/D-m*m,0.f)+1e-5f); }
    __syncthreads();
    for (int i = tid; i < D; i += nt) s_t[i] = (s_o[i]-red[0])*red[1]*e.l2g[i] + e.l2b[i];
    __syncthreads();
  }

  // attn2 -> h row 0
  gemv_sh(P.a2_inw + (size_t)2*D*D, P.a2_inb + 2*D, s_t, s_v, D, D, tid, nt);
  __syncthreads();
  gemv_sh(P.a2_outw, P.a2_outb, s_v, s_o, D, D, tid, nt);
  __syncthreads();
  for (int i = tid; i < D; i += nt) P.h[i] = s_o[i];
  if (tid == 0) { float s=0.f; for (int i=0;i<D;++i) s += s_a1[i]; P.scal[0] = s / (float)D; }
}

// ---------------- h init / degree ----------------
__global__ void init_h_kernel(const float* __restrict__ scores, float* __restrict__ h) {
  size_t idx = (size_t)blockIdx.x*blockDim.x + threadIdx.x;
  if (idx < (size_t)(NN-1)*D) h[(size_t)D + idx] = scores[idx];
}
__global__ void deg_init_kernel(float* deg) {
  int i = blockIdx.x*blockDim.x + threadIdx.x; if (i < NN) deg[i] = 1.0f;  // self-loop
}
__global__ void deg_count_kernel(const int* __restrict__ ei, float* deg) {
  int e = blockIdx.x*blockDim.x + threadIdx.x; if (e < NE) atomicAdd(&deg[ei[NE + e]], 1.0f);
}
__global__ void dis_kernel(float* deg) {
  int i = blockIdx.x*blockDim.x + threadIdx.x;
  if (i < NN) deg[i] = rsqrtf(fmaxf(deg[i], 1e-12f));
}

// ---------------- WMMA GEMM: hw = h @ W.T  (bf16 inputs, f32 accum) ----------------
// Whole 352x352 W staged in LDS as bf16 (247,808 B; CDNA5 WGP has 320 KB).
// B fragments double-buffered 2 k-tiles ahead; k-chain split into two
// independent accumulation chains (even/odd k) for XDL ILP.
__global__ __launch_bounds__(GWAVES*32) void gcn_gemm_kernel(
    const float* __restrict__ h, const float* __restrict__ W, float* __restrict__ hw) {
  extern __shared__ v16bf smemv[];                 // dynamic LDS, 32B aligned
  __bf16* Wl = (__bf16*)smemv;                     // Wl[o*D + k]
  const int tid = threadIdx.x;
  for (int i = tid; i < D*D; i += GWAVES*32) Wl[i] = (__bf16)W[i];
  __syncthreads();

  const int lane = tid & 31, wave = tid >> 5;
  const int hs = lane >> 4;                        // lane half (A/B K-interleave select)
  const int l  = lane & 15;

  for (int mt = blockIdx.x*GWAVES + wave; mt < MTILES; mt += gridDim.x*GWAVES) {
    const int row = mt*16 + l;
    const float* __restrict__ hr = h + (size_t)row * D;
    __builtin_prefetch(hr + (size_t)gridDim.x * GWAVES * 16 * D, 0, 1);

    // A fragments: 16x32 bf16 per k-tile.
    // lanes 0-15: row l, K = kb..kb+15; lanes 16-31: +16.
    v16bf A[KT];
    #pragma unroll
    for (int kt = 0; kt < KT; ++kt) {
      const int kb = kt*32 + hs*16;
      #pragma unroll
      for (int j = 0; j < 8; ++j) {
        A[kt][j]     = (__bf16)hr[kb + j];
        A[kt][8 + j] = (__bf16)hr[kb + 8 + j];
      }
    }

    #pragma unroll 1
    for (int ot = 0; ot < OT; ++ot) {
      const __bf16* wl = Wl + (size_t)(ot*16 + l) * D;   // B col = W row (o)
      v16bf B[2];
      B[0] = *(const v16bf*)(wl + 0*32 + hs*16);         // preload kt=0,1
      B[1] = *(const v16bf*)(wl + 1*32 + hs*16);
      v8f c0 = {0.f,0.f,0.f,0.f,0.f,0.f,0.f,0.f};
      v8f c1 = {0.f,0.f,0.f,0.f,0.f,0.f,0.f,0.f};
      #pragma unroll
      for (int kt = 0; kt < KT; ++kt) {
        v16bf Bc = B[kt & 1];
        if (kt + 2 < KT)                                  // load 2 tiles ahead
          B[kt & 1] = *(const v16bf*)(wl + (kt + 2)*32 + hs*16);
        if (kt & 1)
          c1 = __builtin_amdgcn_wmma_f32_16x16x32_bf16(
                   false, A[kt], false, Bc, (short)0, c1, false, false);
        else
          c0 = __builtin_amdgcn_wmma_f32_16x16x32_bf16(
                   false, A[kt], false, Bc, (short)0, c0, false, false);
      }
      v8f c = c0 + c1;
      // C/D layout: VGPR r -> (M=r, N=l) lanes 0-15; (M=r+8, N=l) lanes 16-31
      float* __restrict__ out = hw + (size_t)mt*16*D + ot*16 + l;
      #pragma unroll
      for (int r = 0; r < 8; ++r) out[(size_t)(r + hs*8) * D] = c[r];
    }
  }
}

// ---------------- GCN aggregate ----------------
// one block per node row: blockIdx = n, threadIdx = feature
__global__ __launch_bounds__(D) void agg_init_kernel(
    const float* __restrict__ hw, const float* __restrict__ dis,
    const float* __restrict__ bias, float* __restrict__ agg) {
  const int n = blockIdx.x, d = threadIdx.x;
  const size_t idx = (size_t)n*D + d;
  const float w = dis[n];
  agg[idx] = hw[idx]*w*w + bias[d];                 // self-loop + bias
}

__global__ __launch_bounds__(D) void edge_scatter_kernel(
    const int* __restrict__ ei, const float* __restrict__ hw,
    const float* __restrict__ dis, float* __restrict__ agg) {
  const int e = blockIdx.x;
  const int s = ei[e], t = ei[NE + e];
  const float w = dis[s]*dis[t];
  const int d = threadIdx.x;
  atomicAdd(&agg[(size_t)t*D + d], hw[(size_t)s*D + d]*w);
}

__global__ __launch_bounds__(128) void ln_gelu_kernel(
    const float* __restrict__ agg, const float* __restrict__ g,
    const float* __restrict__ b, float* __restrict__ hout) {
  __shared__ float rs[128], rs2[128], stat[2];
  const int n = blockIdx.x, tid = threadIdx.x;
  const float* x = agg + (size_t)n*D;
  float s = 0.f, s2 = 0.f;
  for (int d = tid; d < D; d += 128) { float v = x[d]; s += v; s2 += v*v; }
  rs[tid] = s; rs2[tid] = s2;
  __syncthreads();
  if (tid == 0) {
    float a = 0.f, a2 = 0.f;
    for (int i = 0; i < 128; ++i) { a += rs[i]; a2 += rs2[i]; }
    float m = a / D; stat[0] = m; stat[1] = rsqrtf(fmaxf(a2/D - m*m, 0.f) + 1e-5f);
  }
  __syncthreads();
  const float m = stat[0], r = stat[1];
  float* o = hout + (size_t)n*D;
  for (int d = tid; d < D; d += 128) {
    float v = (x[d]-m)*r*g[d] + b[d];
    o[d] = 0.5f*v*(1.0f + erff(v*0.70710678118654752f));   // exact GELU
  }
}

// ---------------- pooling + readout ----------------
__global__ void pooled_zero_kernel(float* pooled) {
  if ((int)threadIdx.x < D) pooled[threadIdx.x] = 0.f;
}
__global__ __launch_bounds__(D) void pool_kernel(const float* __restrict__ h,
                                                 float* __restrict__ pooled) {
  const int d = threadIdx.x;
  int r0 = blockIdx.x * 100, r1 = r0 + 100; if (r1 > NN) r1 = NN;
  float s = 0.f;
  for (int r = r0; r < r1; ++r) s += h[(size_t)r*D + d];
  atomicAdd(&pooled[d], s);
}

struct FinP { const float *m1b,*m1w,*m2b,*m2w,*pooled,*scal; float* out; };
__global__ __launch_bounds__(512) void final_kernel(FinP P) {
  __shared__ float sp[D], sq[D];
  const int tid = threadIdx.x;
  for (int i = tid; i < D; i += 512) sp[i] = P.pooled[i] * (1.0f/(float)NN);
  __syncthreads();
  for (int o = tid; o < D; o += 512) {
    float acc = P.m1b[o]; const float* w = P.m1w + (size_t)o*D;
    for (int i = 0; i < D; ++i) acc = fmaf(sp[i], w[i], acc);
    sq[o] = fmaxf(acc, 0.f);
  }
  __syncthreads();
  const float am = P.scal[0];                       // mean(attn1)
  for (int j = tid; j < INDIM; j += 512) {
    float acc = P.m2b[j]; const float* w = P.m2w + (size_t)j*D;
    for (int i = 0; i < D; ++i) acc = fmaf(sq[i], w[i], acc);
    P.out[j] = am * acc;
  }
}

// ---------------- host launcher ----------------
extern "C" void kernel_launch(void* const* d_in, const int* in_sizes, int n_in,
                              void* d_out, int out_size, void* d_ws, size_t ws_size,
                              hipStream_t stream) {
  (void)in_sizes; (void)n_in; (void)out_size; (void)ws_size;
  const float* state  = (const float*)d_in[0];
  const float* scores = (const float*)d_in[1];
  const int*   eidx   = (const int*)d_in[2];
  auto F = [&](int i) { return (const float*)d_in[i]; };

  // workspace: h reused as agg (GEMM fully consumes h before agg_init runs)
  float* ws     = (float*)d_ws;
  float* h      = ws;                       // NN*D
  float* hw     = h  + (size_t)NN*D;        // NN*D
  float* dis    = hw + (size_t)NN*D;        // NN
  float* attn1v = dis + NN;                 // D
  float* scal   = attn1v + D;               // 2
  float* pooled = scal + 2;                 // D
  float* agg    = h;                        // alias, see ordering above

  FrontP P{};
  P.state = state;
  P.projb = F(IDX_PROJB); P.projw = F(IDX_PROJW);
  P.a1_inb = F(IDX_A1+0); P.a1_inw = F(IDX_A1+1); P.a1_outb = F(IDX_A1+2); P.a1_outw = F(IDX_A1+3);
  P.a2_inb = F(IDX_A2+0); P.a2_inw = F(IDX_A2+1); P.a2_outb = F(IDX_A2+2); P.a2_outw = F(IDX_A2+3);
  for (int l = 0; l < 3; ++l) {
    int b = IDX_ENC(l);
    P.enc[l] = EncP{ F(b+0),F(b+1),F(b+2),F(b+3),F(b+4),F(b+5),
                     F(b+6),F(b+7),F(b+8),F(b+9),F(b+10),F(b+11) };
  }
  P.h = h; P.attn1v = attn1v; P.scal = scal;

  frontend_kernel<<<1, 512, 0, stream>>>(P);
  {
    size_t tot = (size_t)(NN-1)*D;
    init_h_kernel<<<(int)((tot + 255)/256), 256, 0, stream>>>(scores, h);
  }
  deg_init_kernel <<<(NN+255)/256, 256, 0, stream>>>(dis);
  deg_count_kernel<<<(NE+255)/256, 256, 0, stream>>>(eidx, dis);
  dis_kernel      <<<(NN+255)/256, 256, 0, stream>>>(dis);

  const size_t ldsW = (size_t)D * D * 2;    // bf16 W stage: 247,808 B
  for (int l = 0; l < 6; ++l) {
    const float* cb  = F(IDX_CONV(l)+0);
    const float* lnb = F(IDX_CONV(l)+1);
    const float* lng = F(IDX_CONV(l)+2);
    const float* cw  = F(IDX_CONV(l)+3);
    gcn_gemm_kernel<<<391, GWAVES*32, ldsW, stream>>>(h, cw, hw);
    agg_init_kernel<<<NN, D, 0, stream>>>(hw, dis, cb, agg);
    edge_scatter_kernel<<<NE, D, 0, stream>>>(eidx, hw, dis, agg);
    ln_gelu_kernel<<<NN, 128, 0, stream>>>(agg, lng, lnb, h);
  }

  pooled_zero_kernel<<<1, D, 0, stream>>>(pooled);
  pool_kernel<<<500, D, 0, stream>>>(h, pooled);
  FinP FP{ F(IDX_MLP1B), F(IDX_MLP1W), F(IDX_MLP2B), F(IDX_MLP2W), pooled, scal, (float*)d_out };
  final_kernel<<<1, 512, 0, stream>>>(FP);
}
